// HVAE_47674136985983
// MI455X (gfx1250) — compile-verified, matrix operand order checked by reading, not run on hardware
//
#include <hip/hip_runtime.h>

typedef _Float16 v16h __attribute__((ext_vector_type(16)));
typedef _Float16 v8h  __attribute__((ext_vector_type(8)));
typedef float    v8f  __attribute__((ext_vector_type(8)));

#define DEVINL __device__ __forceinline__

// ---------------- problem constants ----------------
#define HH    128   // hidden
#define OOW   32    // output / input width of GRUs
#define IIW   32    // z width
#define MR    32    // batch rows per workgroup
#define MAXD  5

// ---------------- fast transcendentals (hardware rcp/exp) ----------------
DEVINL float frcp(float x)  { return __builtin_amdgcn_rcpf(x); }
DEVINL float fsig(float x)  { return frcp(1.0f + __expf(-x)); }
DEVINL float ftanh(float x) { return 1.0f - 2.0f * frcp(1.0f + __expf(2.0f * x)); }

// ---------------- packed-weight workspace offsets (f16 elements) ----------------
enum : int {
  OFF_Z2H    = 0,
  OFF_H2O    = OFF_Z2H   + IIW * HH,        // 32x128
  OFF_TEST   = OFF_H2O   + HH * OOW,        // 128x32
  OFF_TEST2  = OFF_TEST  + HH * HH,
  OFF_TEST3  = OFF_TEST2 + HH * HH,
  OFF_ANC_WI = OFF_TEST3 + HH * HH,         // 3 x 32x128
  OFF_ANC_WH = OFF_ANC_WI + 3 * OOW * HH,   // 3 x 128x128
  OFF_FRA_WI = OFF_ANC_WH + 3 * HH * HH,
  OFF_FRA_WH = OFF_FRA_WI + 3 * OOW * HH,
  OFF_G_WI   = OFF_FRA_WH + 3 * HH * HH,
  OFF_G_WH   = OFF_G_WI  + 3 * OOW * HH,    // 3 x 256x128
  WS_F16_TOTAL = OFF_G_WH + 3 * 2 * HH * HH
};

// =====================================================================
// Pack a row-major f32 weight W[K][N] into WMMA-B-fragment order, f16.
// Tile t = ntile*(K/32)+kc holds 512 halves: [lane 0..31][e 0..15].
//   lane<16 : n = ntile*16+lane,    k = kc*32 + (e<8 ? e    : 8+e)
//   lane>=16: n = ntile*16+lane-16, k = kc*32 + (e<8 ? 8+e : 16+e)
// =====================================================================
__global__ void hvae_pack_w(const float* __restrict__ src,
                            _Float16* __restrict__ dst, int K, int N) {
  int i = blockIdx.x * blockDim.x + threadIdx.x;
  if (i >= K * N) return;
  int e    = i & 15;
  int lane = (i >> 4) & 31;
  int t    = i >> 9;
  int kcTotal = K >> 5;
  int ntile = t / kcTotal;
  int kc    = t % kcTotal;
  int n = ntile * 16 + (lane & 15);
  int k;
  if (lane < 16) k = kc * 32 + ((e < 8) ? e     : 8 + e);
  else           k = kc * 32 + ((e < 8) ? 8 + e : 16 + e);
  dst[i] = (_Float16)src[(size_t)k * N + n];
}

// =====================================================================
// Register accumulation core: acc(0|1) += A16[32][AST] x Wpk(kc window).
// acc0 covers rows mb..mb+7, acc1 rows 16+mb..16+mb+7 of column ncol.
// =====================================================================
DEVINL void accum_pair(const _Float16* __restrict__ A16, int AST,
                       const _Float16* __restrict__ Wpk,
                       int kcBase, int kcCount, int kcTotal,
                       int w, int lane, v8f& acc0, v8f& acc1) {
  const int arow   = lane & 15;
  const int klhalf = (lane & 16) ? 8 : 0;
  for (int kci = 0; kci < kcCount; ++kci) {
    const v16h bfrag =
        *(const v16h*)(Wpk + (((size_t)(w * kcTotal + kcBase + kci)) << 9) + (lane << 4));
    const int koff = kci * 32 + klhalf;
    const _Float16* ap0 = A16 + arow * AST + koff;
    v8h lo0 = *(const v8h*)ap0;
    v8h hi0 = *(const v8h*)(ap0 + 16);
    const _Float16* ap1 = A16 + (16 + arow) * AST + koff;
    v8h lo1 = *(const v8h*)ap1;
    v8h hi1 = *(const v8h*)(ap1 + 16);
    v16h a0, a1;
#pragma unroll
    for (int q = 0; q < 8; ++q) {
      a0[q] = lo0[q]; a0[q + 8] = hi0[q];
      a1[q] = lo1[q]; a1[q + 8] = hi1[q];
    }
    acc0 = __builtin_amdgcn_wmma_f32_16x16x32_f16(false, a0, false, bfrag,
                                                  (short)0, acc0, false, false);
    acc1 = __builtin_amdgcn_wmma_f32_16x16x32_f16(false, a1, false, bfrag,
                                                  (short)0, acc1, false, false);
  }
}

// =====================================================================
// Wave-tiled WMMA GEMM with LDS output (used for plain linear layers).
// =====================================================================
template <int AST>
DEVINL void wgemm(const _Float16* __restrict__ A16,
                  const _Float16* __restrict__ Wpk,
                  int kcCount, int nTiles,
                  float* __restrict__ D, _Float16* __restrict__ D16,
                  const float* __restrict__ bias) {
  const int w    = threadIdx.x >> 5;
  const int lane = threadIdx.x & 31;
  if (w >= nTiles) return;
  const int NW   = nTiles * 16;
  const int ncol = w * 16 + (lane & 15);
  const int mb   = (lane & 16) ? 8 : 0;
  const float bv = bias ? bias[ncol] : 0.0f;

  v8f acc0, acc1;
#pragma unroll
  for (int r = 0; r < 8; ++r) { acc0[r] = bv; acc1[r] = bv; }
  accum_pair(A16, AST, Wpk, 0, kcCount, kcCount, w, lane, acc0, acc1);

#pragma unroll
  for (int r = 0; r < 8; ++r) {
    if (D) {
      D[(mb + r) * NW + ncol]      = acc0[r];
      D[(16 + mb + r) * NW + ncol] = acc1[r];
    }
    if (D16) {
      D16[(mb + r) * NW + ncol]      = (_Float16)acc0[r];
      D16[(16 + mb + r) * NW + ncol] = (_Float16)acc1[r];
    }
  }
}

DEVINL v8f sigv(v8f a) {
  v8f o;
#pragma unroll
  for (int r = 0; r < 8; ++r) o[r] = fsig(a[r]);
  return o;
}

// =====================================================================
// Fully register-resident GRU cell over the 32-row tile. N=128 -> all
// 8 waves active. No internal barriers; dst16 must not alias x16/h16.
//   gates [r,z,n]; n = tanh(x@wi2+bi2 + r*(h@wh2+bh2));
//   h' = (1-z)*n + z*h
// =====================================================================
DEVINL void gru_std_reg(const _Float16* __restrict__ x16,
                        const _Float16* __restrict__ h16,
                        const _Float16* __restrict__ WI,
                        const _Float16* __restrict__ WH,
                        const float* __restrict__ BI,
                        const float* __restrict__ BH,
                        _Float16* __restrict__ dst16) {
  const int w    = threadIdx.x >> 5;
  const int lane = threadIdx.x & 31;
  const int ncol = w * 16 + (lane & 15);
  const int mb   = (lane & 16) ? 8 : 0;

  // r gate
  float b = BI[0 * HH + ncol] + BH[0 * HH + ncol];
  v8f r0, r1;
#pragma unroll
  for (int r = 0; r < 8; ++r) { r0[r] = b; r1[r] = b; }
  accum_pair(x16, OOW, WI + 0 * OOW * HH, 0, 1, 1, w, lane, r0, r1);
  accum_pair(h16, HH,  WH + 0 * HH * HH,  0, 4, 4, w, lane, r0, r1);
  r0 = sigv(r0); r1 = sigv(r1);

  // z gate
  b = BI[1 * HH + ncol] + BH[1 * HH + ncol];
  v8f z0, z1;
#pragma unroll
  for (int r = 0; r < 8; ++r) { z0[r] = b; z1[r] = b; }
  accum_pair(x16, OOW, WI + 1 * OOW * HH, 0, 1, 1, w, lane, z0, z1);
  accum_pair(h16, HH,  WH + 1 * HH * HH,  0, 4, 4, w, lane, z0, z1);
  z0 = sigv(z0); z1 = sigv(z1);

  // n gate: gh part first, multiply by r in regs, then accumulate gi part
  float bh2 = BH[2 * HH + ncol];
  float bi2 = BI[2 * HH + ncol];
  v8f n0, n1;
#pragma unroll
  for (int r = 0; r < 8; ++r) { n0[r] = bh2; n1[r] = bh2; }
  accum_pair(h16, HH, WH + 2 * HH * HH, 0, 4, 4, w, lane, n0, n1);
#pragma unroll
  for (int r = 0; r < 8; ++r) {
    n0[r] = n0[r] * r0[r] + bi2;
    n1[r] = n1[r] * r1[r] + bi2;
  }
  accum_pair(x16, OOW, WI + 2 * OOW * HH, 0, 1, 1, w, lane, n0, n1);

  // blend + store (each thread touches only its own 16 elements)
#pragma unroll
  for (int r = 0; r < 8; ++r) {
    int row0 = mb + r, row1 = 16 + mb + r;
    float h0 = (float)h16[row0 * HH + ncol];
    float h1 = (float)h16[row1 * HH + ncol];
    float o0 = (1.0f - z0[r]) * ftanh(n0[r]) + z0[r] * h0;
    float o1 = (1.0f - z1[r]) * ftanh(n1[r]) + z1[r] * h1;
    dst16[row0 * HH + ncol] = (_Float16)o0;
    dst16[row1 * HH + ncol] = (_Float16)o1;
  }
}

// =====================================================================
// Register-resident GRU221: gates see concat(h1,h2) (K=256);
// out = (1-z)*n + a1*h1 + a2*h2. One internal barrier (a1/a2 fan-out).
// dst16 must not alias h1/h2/xg.
// =====================================================================
DEVINL void gru221_reg(const _Float16* __restrict__ xg,
                       const _Float16* __restrict__ h1,
                       const _Float16* __restrict__ h2,
                       const _Float16* __restrict__ WI,
                       const _Float16* __restrict__ WH,
                       const float* __restrict__ BI,
                       const float* __restrict__ BH,
                       const float* __restrict__ W1, const float* __restrict__ B1,
                       const float* __restrict__ W2, const float* __restrict__ B2,
                       _Float16* __restrict__ dst16,
                       float* __restrict__ A1, float* __restrict__ A2) {
  const int tid  = threadIdx.x;
  const int w    = tid >> 5;
  const int lane = tid & 31;
  const int ncol = w * 16 + (lane & 15);
  const int mb   = (lane & 16) ? 8 : 0;
  const int gWH  = 8 * OOW * HH;  // 256x128 per gate

  // per-row scalar gates a1 = h1@w1 + b1, a2 = h2@w2 + b2
  // 8 lanes per row, 16 elements each, width-8 shuffle reduction
  {
    const int row = tid >> 3;
    const int sub = tid & 7;
    float s1 = 0.0f, s2 = 0.0f;
#pragma unroll
    for (int j = 0; j < 16; ++j) {
      int k = sub * 16 + j;
      s1 += (float)h1[row * HH + k] * W1[k];
      s2 += (float)h2[row * HH + k] * W2[k];
    }
#pragma unroll
    for (int m = 1; m < 8; m <<= 1) {
      s1 += __shfl_xor(s1, m, 8);
      s2 += __shfl_xor(s2, m, 8);
    }
    if (sub == 0) {
      A1[row] = s1 + B1[0];
      A2[row] = s2 + B2[0];
    }
  }
  __syncthreads();

  // r gate
  float b = BI[0 * HH + ncol] + BH[0 * HH + ncol];
  v8f r0, r1;
#pragma unroll
  for (int r = 0; r < 8; ++r) { r0[r] = b; r1[r] = b; }
  accum_pair(xg, OOW, WI + 0 * OOW * HH, 0, 1, 1, w, lane, r0, r1);
  accum_pair(h1, HH,  WH + 0 * gWH, 0, 4, 8, w, lane, r0, r1);
  accum_pair(h2, HH,  WH + 0 * gWH, 4, 4, 8, w, lane, r0, r1);
  r0 = sigv(r0); r1 = sigv(r1);

  // z gate
  b = BI[1 * HH + ncol] + BH[1 * HH + ncol];
  v8f z0, z1;
#pragma unroll
  for (int r = 0; r < 8; ++r) { z0[r] = b; z1[r] = b; }
  accum_pair(xg, OOW, WI + 1 * OOW * HH, 0, 1, 1, w, lane, z0, z1);
  accum_pair(h1, HH,  WH + 1 * gWH, 0, 4, 8, w, lane, z0, z1);
  accum_pair(h2, HH,  WH + 1 * gWH, 4, 4, 8, w, lane, z0, z1);
  z0 = sigv(z0); z1 = sigv(z1);

  // n gate
  float bh2 = BH[2 * HH + ncol];
  float bi2 = BI[2 * HH + ncol];
  v8f n0, n1;
#pragma unroll
  for (int r = 0; r < 8; ++r) { n0[r] = bh2; n1[r] = bh2; }
  accum_pair(h1, HH, WH + 2 * gWH, 0, 4, 8, w, lane, n0, n1);
  accum_pair(h2, HH, WH + 2 * gWH, 4, 4, 8, w, lane, n0, n1);
#pragma unroll
  for (int r = 0; r < 8; ++r) {
    n0[r] = n0[r] * r0[r] + bi2;
    n1[r] = n1[r] * r1[r] + bi2;
  }
  accum_pair(xg, OOW, WI + 2 * OOW * HH, 0, 1, 1, w, lane, n0, n1);

  // blend + store
#pragma unroll
  for (int r = 0; r < 8; ++r) {
    int row0 = mb + r, row1 = 16 + mb + r;
    float h10 = (float)h1[row0 * HH + ncol];
    float h11 = (float)h1[row1 * HH + ncol];
    float h20 = (float)h2[row0 * HH + ncol];
    float h21 = (float)h2[row1 * HH + ncol];
    float o0 = (1.0f - z0[r]) * ftanh(n0[r]) + A1[row0] * h10 + A2[row0] * h20;
    float o1 = (1.0f - z1[r]) * ftanh(n1[r]) + A1[row1] * h11 + A2[row1] * h21;
    dst16[row0 * HH + ncol] = (_Float16)o0;
    dst16[row1 * HH + ncol] = (_Float16)o1;
  }
}

// =====================================================================
// Persistent per-batch-tile tree walker. One workgroup owns 32 batch
// rows and walks the full preorder tree with an explicit LDS stack.
// =====================================================================
struct HParams {
  const float* z;
  const float* z2h_b; const float* h2o_b;
  const float* test_b; const float* test2_b; const float* test3_b;
  const float* anc_bi; const float* anc_bh;
  const float* fra_bi; const float* fra_bh;
  const float* g_bi;   const float* g_bh;
  const float* g_w1;   const float* g_b1;
  const float* g_w2;   const float* g_b2;
  const int*   depth;
  const _Float16* ws;
  float* out;
};

__global__ __launch_bounds__(256) void hvae_tree_kernel(HParams p) {
  __shared__ float sPred[MR * OOW];
  __shared__ float sA1[MR], sA2[MR];
  __shared__ __align__(32) _Float16 sHA[MR * HH];   // current hidden_a
  __shared__ __align__(32) _Float16 sHD[MR * HH];   // current hidden / hidden_f
  __shared__ __align__(32) _Float16 sHF[MR * HH];   // pre-GRU hidden_f / hidden_new / hidden0
  __shared__ __align__(32) _Float16 sSP[MR * OOW];  // current softmax
  __shared__ __align__(32) _Float16 sR[MR * OOW];   // child return value (sp_f)
  __shared__ __align__(32) _Float16 sXG[MR * OOW];  // (sp+sp_f)/2 ; also z staging
  __shared__ __align__(32) _Float16 sFrH[MAXD][MR * HH];   // stack: h_ai
  __shared__ __align__(32) _Float16 sFrS[MAXD][MR * OOW];  // stack: sp

  const int tid  = threadIdx.x;
  const int base = blockIdx.x * MR;
  const int Btot = gridDim.x * MR;
  const _Float16* ws = p.ws;
  int dpt = p.depth[0];
  if (dpt > MAXD) dpt = MAXD;
  if (dpt < 0) dpt = 0;

  // hidden0 = z @ z2h + b ; hidden_a = hidden0@test2+b ; hidden = hidden0@test3+b
  for (int i = tid; i < MR * IIW; i += 256)
    sXG[i] = (_Float16)p.z[(size_t)(base + (i >> 5)) * IIW + (i & 31)];
  __syncthreads();
  wgemm<IIW>(sXG, ws + OFF_Z2H, 1, 8, nullptr, sHF, p.z2h_b);
  __syncthreads();
  wgemm<HH>(sHF, ws + OFF_TEST2, 4, 8, nullptr, sHA, p.test2_b);
  __syncthreads();
  wgemm<HH>(sHF, ws + OFF_TEST3, 4, 8, nullptr, sHD, p.test3_b);
  __syncthreads();

  int stage[MAXD];
  int node = 0, level = 0;
  bool visiting = true;
  for (;;) {
    if (visiting) {
      // prediction = hidden @ h2o + b  -> emit logits (preorder node index)
      wgemm<HH>(sHD, ws + OFF_H2O, 4, 2, sPred, nullptr, p.h2o_b);
      __syncthreads();
      for (int i = tid; i < MR * OOW; i += 256)
        p.out[((size_t)node * Btot + base + (i >> 5)) * OOW + (i & 31)] = sPred[i];
      // softmax over O=32: 8 lanes per row, 4 cols each, shuffle reductions
      {
        const int row = tid >> 3;
        const int sub = tid & 7;
        const float* pr = sPred + row * OOW + sub * 4;
        float v0 = pr[0], v1 = pr[1], v2 = pr[2], v3 = pr[3];
        float mx = fmaxf(fmaxf(v0, v1), fmaxf(v2, v3));
#pragma unroll
        for (int m = 1; m < 8; m <<= 1) mx = fmaxf(mx, __shfl_xor(mx, m, 8));
        v0 = __expf(v0 - mx); v1 = __expf(v1 - mx);
        v2 = __expf(v2 - mx); v3 = __expf(v3 - mx);
        float s = v0 + v1 + v2 + v3;
#pragma unroll
        for (int m = 1; m < 8; m <<= 1) s += __shfl_xor(s, m, 8);
        float inv = frcp(s);
        _Float16* sp = sSP + row * OOW + sub * 4;
        sp[0] = (_Float16)(v0 * inv); sp[1] = (_Float16)(v1 * inv);
        sp[2] = (_Float16)(v2 * inv); sp[3] = (_Float16)(v3 * inv);
      }
      __syncthreads();
      ++node;
      if (level < dpt) {
        // h_ai = GRU_anc(sp, hidden_a) -> stack frame (register-resident GRU)
        gru_std_reg(sSP, sHA, ws + OFF_ANC_WI, ws + OFF_ANC_WH,
                    p.anc_bi, p.anc_bh, sFrH[level]);
        for (int i = tid; i < MR * OOW; i += 256) sFrS[level][i] = sSP[i];
        __syncthreads();
        stage[level] = 0;
        // descend into first child
        wgemm<HH>(sFrH[level], ws + OFF_TEST2, 4, 8, nullptr, sHA, p.test2_b);
        __syncthreads();
        wgemm<HH>(sFrH[level], ws + OFF_TEST3, 4, 8, nullptr, sHD, p.test3_b);
        __syncthreads();
        ++level;
      } else {
        // leaf: return sp (anc GRU result is dead at leaves -> skipped)
        for (int i = tid; i < MR * OOW; i += 256) sR[i] = sSP[i];
        __syncthreads();
        visiting = false;
      }
    } else {
      if (level == 0) break;
      --level;
      _Float16* frH = sFrH[level];
      _Float16* frS = sFrS[level];
      if (stage[level] == 0) {
        stage[level] = 1;
        // hidden_f = GRU_fra(sp_f, h_ai@test+b)   (result -> sHD, dead here)
        wgemm<HH>(frH, ws + OFF_TEST, 4, 8, nullptr, sHF, p.test_b);
        __syncthreads();
        gru_std_reg(sR, sHF, ws + OFF_FRA_WI, ws + OFF_FRA_WH,
                    p.fra_bi, p.fra_bh, sHD);
        // hidden_new = GRU221((sp+sp_f)/2, h_ai, hidden_f)  (result -> sHF)
        for (int i = tid; i < MR * OOW; i += 256)
          sXG[i] = (_Float16)(0.5f * ((float)frS[i] + (float)sR[i]));
        __syncthreads();
        gru221_reg(sXG, frH, sHD, ws + OFF_G_WI, ws + OFF_G_WH, p.g_bi, p.g_bh,
                   p.g_w1, p.g_b1, p.g_w2, p.g_b2, sHF, sA1, sA2);
        __syncthreads();
        // descend into second child
        wgemm<HH>(frH, ws + OFF_TEST2, 4, 8, nullptr, sHA, p.test2_b);
        __syncthreads();
        wgemm<HH>(sHF, ws + OFF_TEST3, 4, 8, nullptr, sHD, p.test3_b);
        __syncthreads();
        ++level;
        visiting = true;
      } else {
        // finished second child: restore our own sp as the return value
        for (int i = tid; i < MR * OOW; i += 256) sR[i] = frS[i];
        __syncthreads();
      }
    }
  }
}

// =====================================================================
extern "C" void kernel_launch(void* const* d_in, const int* in_sizes, int n_in,
                              void* d_out, int out_size, void* d_ws, size_t ws_size,
                              hipStream_t stream) {
  (void)n_in; (void)out_size; (void)ws_size;
  const float* z       = (const float*)d_in[0];
  const float* z2h_w   = (const float*)d_in[1];
  const float* z2h_b   = (const float*)d_in[2];
  const float* h2o_w   = (const float*)d_in[3];
  const float* h2o_b   = (const float*)d_in[4];
  const float* test_w  = (const float*)d_in[5];
  const float* test_b  = (const float*)d_in[6];
  const float* test2_w = (const float*)d_in[7];
  const float* test2_b = (const float*)d_in[8];
  const float* test3_w = (const float*)d_in[9];
  const float* test3_b = (const float*)d_in[10];
  const float* anc_wi  = (const float*)d_in[11];
  const float* anc_bi  = (const float*)d_in[12];
  const float* anc_wh  = (const float*)d_in[13];
  const float* anc_bh  = (const float*)d_in[14];
  const float* fra_wi  = (const float*)d_in[15];
  const float* fra_bi  = (const float*)d_in[16];
  const float* fra_wh  = (const float*)d_in[17];
  const float* fra_bh  = (const float*)d_in[18];
  const float* g_wi    = (const float*)d_in[19];
  const float* g_bi    = (const float*)d_in[20];
  const float* g_wh    = (const float*)d_in[21];
  const float* g_bh    = (const float*)d_in[22];
  const float* g_w1    = (const float*)d_in[23];
  const float* g_b1    = (const float*)d_in[24];
  const float* g_w2    = (const float*)d_in[25];
  const float* g_b2    = (const float*)d_in[26];
  const int*   depth   = (const int*)d_in[27];

  _Float16* ws = (_Float16*)d_ws;

  auto pack = [&](const float* src, int off, int K, int N) {
    int tot = K * N;
    hvae_pack_w<<<(tot + 255) / 256, 256, 0, stream>>>(src, ws + off, K, N);
  };
  pack(z2h_w,  OFF_Z2H,  IIW, HH);
  pack(h2o_w,  OFF_H2O,  HH,  OOW);
  pack(test_w,  OFF_TEST,  HH, HH);
  pack(test2_w, OFF_TEST2, HH, HH);
  pack(test3_w, OFF_TEST3, HH, HH);
  for (int g = 0; g < 3; ++g) {
    pack(anc_wi + g * IIW * HH, OFF_ANC_WI + g * IIW * HH, IIW, HH);
    pack(anc_wh + g * HH * HH,  OFF_ANC_WH + g * HH * HH,  HH,  HH);
    pack(fra_wi + g * IIW * HH, OFF_FRA_WI + g * IIW * HH, IIW, HH);
    pack(fra_wh + g * HH * HH,  OFF_FRA_WH + g * HH * HH,  HH,  HH);
    pack(g_wi + g * IIW * HH,   OFF_G_WI + g * IIW * HH,   IIW, HH);
    pack(g_wh + g * 2 * HH * HH, OFF_G_WH + g * 2 * HH * HH, 2 * HH, HH);
  }

  HParams p;
  p.z = z; p.z2h_b = z2h_b; p.h2o_b = h2o_b;
  p.test_b = test_b; p.test2_b = test2_b; p.test3_b = test3_b;
  p.anc_bi = anc_bi; p.anc_bh = anc_bh;
  p.fra_bi = fra_bi; p.fra_bh = fra_bh;
  p.g_bi = g_bi; p.g_bh = g_bh;
  p.g_w1 = g_w1; p.g_b1 = g_b1; p.g_w2 = g_w2; p.g_b2 = g_b2;
  p.depth = depth; p.ws = ws; p.out = (float*)d_out;

  int B = in_sizes[0] / IIW;       // 4096
  int nblocks = B / MR;            // 128 workgroups, 8 wave32s each
  hvae_tree_kernel<<<nblocks, 256, 0, stream>>>(p);
}
